// MLPResNetBlock_6975026888973
// MI455X (gfx1250) — matrix operand order repaired and checked by reading, not compiled, not run on hardware
//
#include <hip/hip_runtime.h>
#include <cstdint>

#define DIM_C 1024
#define B_SZ 4
#define T_SZ 2048
#define H_N 8
#define D_H 128
#define SP 2176            // padded rows per batch: 17*128
#define SKV 2160           // valid kv rows per batch
#define MTOT (B_SZ * SP)   // 8704 = 68 * 128

typedef __attribute__((ext_vector_type(16))) __bf16 v16bf;
typedef __attribute__((ext_vector_type(8)))  __bf16 v8bf;
typedef __attribute__((ext_vector_type(8)))  float  v8f;

// ---------------------------------------------------------------------------
// CDNA5 async global->LDS copy (ASYNCcnt) with synchronous fallback
// Builtin signature (from hipcc diagnostic): param1 = int __vector(4) AS(1)*,
// 4 args total.
// ---------------------------------------------------------------------------
typedef int i4v __attribute__((vector_size(16)));
typedef __attribute__((address_space(1))) i4v* gp_i4;
typedef __attribute__((address_space(3))) i4v* lp_i4;

#if defined(__has_builtin)
#if __has_builtin(__builtin_amdgcn_global_load_async_to_lds_b128)
#define HAVE_ASYNC 1
#endif
#endif

#ifdef HAVE_ASYNC
#define CP16(g, l)                                                   \
  __builtin_amdgcn_global_load_async_to_lds_b128((gp_i4)(g), (lp_i4)(l), 0, 0)
#if __has_builtin(__builtin_amdgcn_s_wait_asynccnt)
#define CP_WAIT() __builtin_amdgcn_s_wait_asynccnt(0)
#else
#define CP_WAIT() asm volatile("s_wait_asynccnt 0x0" ::: "memory")
#endif
#else
#define CP16(g, l) (*(uint4*)(l) = *(const uint4*)(g))
#define CP_WAIT()
#endif

__device__ __forceinline__ v8f wmma_bf16(v16bf a, v16bf b, v8f c) {
  return __builtin_amdgcn_wmma_f32_16x16x32_bf16(false, a, false, b, (short)0, c,
                                                 false, false);
}

// A-fragment (16x32 bf16, row-major source): lane = M row, K split per ISA
// (lanes 0-15: K 0-7 & 16-23; lanes 16-31: K 8-15 & 24-31)
__device__ __forceinline__ v16bf frag_a(const __bf16* base, int lane, int rstride) {
  const __bf16* p = base + (size_t)(lane & 15) * rstride + ((lane >> 4) << 3);
  v8bf lo = *(const v8bf*)(p);
  v8bf hi = *(const v8bf*)(p + 16);
  return __builtin_shufflevector(lo, hi, 0,1,2,3,4,5,6,7,8,9,10,11,12,13,14,15);
}

// B-fragment (32x16 bf16) from an N-major (i.e. B^T row-major) source:
// lane = N column, K split per ISA (lanes 0-15: K 0-7 & 16-23; lanes 16-31: K 8-15 & 24-31)
__device__ __forceinline__ v16bf frag_b(const __bf16* base, int lane, int rstride) {
  const __bf16* p = base + (size_t)(lane & 15) * rstride + ((lane >> 4) << 3);
  v8bf lo = *(const v8bf*)(p);
  v8bf hi = *(const v8bf*)(p + 16);
  return __builtin_shufflevector(lo, hi, 0,1,2,3,4,5,6,7,8,9,10,11,12,13,14,15);
}

__device__ __forceinline__ float redmax16(float v) {
  v = fmaxf(v, __shfl_xor(v, 1));
  v = fmaxf(v, __shfl_xor(v, 2));
  v = fmaxf(v, __shfl_xor(v, 4));
  v = fmaxf(v, __shfl_xor(v, 8));
  return v;
}
__device__ __forceinline__ float redsum16(float v) {
  v += __shfl_xor(v, 1);
  v += __shfl_xor(v, 2);
  v += __shfl_xor(v, 4);
  v += __shfl_xor(v, 8);
  return v;
}

// ---------------------------------------------------------------------------
// Pack [x; h_t; h_a; p] -> bf16 XC[B][SP][1024], pad rows zeroed
// ---------------------------------------------------------------------------
__global__ __launch_bounds__(256) void pack_kernel(
    const float* __restrict__ x, const float* __restrict__ ht,
    const float* __restrict__ ha, const float* __restrict__ pp,
    __bf16* __restrict__ xc) {
  int row = blockIdx.x;
  int b = row / SP, t = row % SP;
  const float* src = nullptr;
  if (t < 2048)      src = x  + ((size_t)(b * 2048 + t)) * DIM_C;
  else if (t < 2112) src = ht + ((size_t)(b * 64 + (t - 2048))) * DIM_C;
  else if (t < 2128) src = ha + ((size_t)(b * 16 + (t - 2112))) * DIM_C;
  else if (t < 2160) src = pp + ((size_t)(b * 32 + (t - 2128))) * DIM_C;
  __bf16* dst = xc + (size_t)row * DIM_C;
  for (int i = threadIdx.x; i < DIM_C; i += 256)
    dst[i] = src ? (__bf16)src[i] : (__bf16)0.0f;
}

// ---------------------------------------------------------------------------
// W[k][n] f32  ->  WT[n][k] bf16  (32x32 LDS tile transpose)
// ---------------------------------------------------------------------------
__global__ __launch_bounds__(256) void wt_kernel(const float* __restrict__ W,
                                                 __bf16* __restrict__ WT) {
  __shared__ float tile[32][33];
  int bi = blockIdx.y * 32, bj = blockIdx.x * 32;
  int tx = threadIdx.x & 31, ty = threadIdx.x >> 5;
#pragma unroll
  for (int i = 0; i < 4; ++i)
    tile[ty + i * 8][tx] = W[(size_t)(bi + ty + i * 8) * DIM_C + bj + tx];
  __syncthreads();
#pragma unroll
  for (int i = 0; i < 4; ++i)
    WT[(size_t)(bj + ty + i * 8) * DIM_C + bi + tx] = (__bf16)tile[tx][ty + i * 8];
}

// ---------------------------------------------------------------------------
// Tiled WMMA GEMM:  C[M=8704][N=1024] = A[M][1024] @ WT[N][1024]^T + bias
// Async global->LDS double-buffered pipeline.
// mode 0: bf16 out (Q/K)      mode 1: bf16 out transposed per head (V^T)
// mode 2: f32 y = acc+bias+resid (valid rows only)
// mode 3: f32 relu(acc+bias)  (valid rows only)
// ---------------------------------------------------------------------------
__global__ __launch_bounds__(256) void gemm_kernel(
    const __bf16* __restrict__ A, const __bf16* __restrict__ WT,
    const float* __restrict__ bias, int mode,
    __bf16* __restrict__ outb, float* __restrict__ outf,
    const float* __restrict__ resid) {
  __shared__ __align__(16) __bf16 ta[2][128 * 32];
  __shared__ __align__(16) __bf16 tb[2][128 * 32];

  int tid = threadIdx.x, lane = tid & 31, wv = tid >> 5;
  int m0 = blockIdx.y * 128, n0 = blockIdx.x * 128;
  int wm = (wv & 3) * 32, wn = (wv >> 2) * 64;

  const __bf16* Ab = A  + (size_t)m0 * DIM_C;
  const __bf16* Wb = WT + (size_t)n0 * DIM_C;

  const int row = tid >> 1;        // 0..127
  const int seg = (tid & 1) * 16;  // halves within 32-wide row

  v8f zero = {0.f, 0.f, 0.f, 0.f, 0.f, 0.f, 0.f, 0.f};
  v8f acc[2][4];
#pragma unroll
  for (int i = 0; i < 2; ++i)
#pragma unroll
    for (int j = 0; j < 4; ++j) acc[i][j] = zero;

  // per-thread: 2x16B for A tile + 2x16B for B tile per k-chunk
  const __bf16* gar = Ab + (size_t)row * DIM_C + seg;
  const __bf16* gbr = Wb + (size_t)row * DIM_C + seg;
  __bf16* lar0 = ta[0] + row * 32 + seg;
  __bf16* lbr0 = tb[0] + row * 32 + seg;
  __bf16* lar1 = ta[1] + row * 32 + seg;
  __bf16* lbr1 = tb[1] + row * 32 + seg;

  CP16(gar, lar0);      CP16(gar + 8, lar0 + 8);
  CP16(gbr, lbr0);      CP16(gbr + 8, lbr0 + 8);

  int buf = 0;
  for (int kc = 0; kc < 32; ++kc) {
    CP_WAIT();           // my async copies into ta/tb[buf] are done
    __syncthreads();     // everyone's copies visible; prev math on buf^1 done
    if (kc < 31) {
      const __bf16* ga = gar + (kc + 1) * 32;
      const __bf16* gb = gbr + (kc + 1) * 32;
      __bf16* la = buf ? lar0 : lar1;
      __bf16* lb = buf ? lbr0 : lbr1;
      CP16(ga, la);      CP16(ga + 8, la + 8);
      CP16(gb, lb);      CP16(gb + 8, lb + 8);
    }
    v16bf af[2], bfr[4];
#pragma unroll
    for (int i = 0; i < 2; ++i)
      af[i] = frag_a(&ta[buf][(wm + i * 16) * 32], lane, 32);
#pragma unroll
    for (int j = 0; j < 4; ++j)
      bfr[j] = frag_b(&tb[buf][(wn + j * 16) * 32], lane, 32);
#pragma unroll
    for (int i = 0; i < 2; ++i)
#pragma unroll
      for (int j = 0; j < 4; ++j)
        acc[i][j] = wmma_bf16(af[i], bfr[j], acc[i][j]);
    buf ^= 1;
  }

  int lcol = lane & 15, lhalf = lane >> 4;
#pragma unroll
  for (int i = 0; i < 2; ++i) {
#pragma unroll
    for (int r = 0; r < 8; ++r) {
      int gm = m0 + wm + i * 16 + r + 8 * lhalf;
      int bb = gm / SP, tt = gm % SP;
#pragma unroll
      for (int j = 0; j < 4; ++j) {
        int gn = n0 + wn + j * 16 + lcol;
        float v = acc[i][j][r] + bias[gn];
        if (mode == 0) {
          outb[(size_t)gm * DIM_C + gn] = (__bf16)v;
        } else if (mode == 1) {
          int hh = gn >> 7, dd = gn & 127;
          outb[(((size_t)(bb * H_N + hh)) * D_H + dd) * SP + tt] = (__bf16)v;
        } else if (mode == 2) {
          if (tt < T_SZ) {
            size_t o = ((size_t)(bb * T_SZ + tt)) * DIM_C + gn;
            outf[o] = v + resid[o];
          }
        } else {
          if (tt < T_SZ) {
            size_t o = ((size_t)(bb * T_SZ + tt)) * DIM_C + gn;
            outf[o] = fmaxf(v, 0.0f);
          }
        }
      }
    }
  }
}

// ---------------------------------------------------------------------------
// Flash attention: block = (b, h, 64 q rows), 4 waves x 16 q rows.
// K/V streamed in 32-key chunks via async double-buffered LDS tiles.
// ---------------------------------------------------------------------------
__global__ __launch_bounds__(128) void attn_kernel(
    const __bf16* __restrict__ Qb, const __bf16* __restrict__ Kb,
    const __bf16* __restrict__ Vt, __bf16* __restrict__ Ob,
    const float* __restrict__ gating) {
  __shared__ __align__(16) __bf16 tk[2][32 * 128];   // [key][d]
  __shared__ __align__(16) __bf16 tv[2][128 * 32];   // [d][key]  (from V^T)
  __shared__ __align__(16) __bf16 pl[4][16 * 32];    // per-wave P scratch

  int tid = threadIdx.x, lane = tid & 31, wv = tid >> 5;
  int b = blockIdx.z, h = blockIdx.y;
  int q0 = blockIdx.x * 64;
  float ratio = tanhf(gating[0]);
  const float scl = 0.08838834764831845f;  // 1/sqrt(128)

  // Q fragments, register resident
  const __bf16* qrow = Qb + ((size_t)(b * SP + q0 + wv * 16)) * DIM_C + h * D_H;
  v16bf qf[4];
#pragma unroll
  for (int kk = 0; kk < 4; ++kk) qf[kk] = frag_a(qrow + kk * 32, lane, DIM_C);

  v8f zero = {0.f, 0.f, 0.f, 0.f, 0.f, 0.f, 0.f, 0.f};
  v8f oacc[8];
#pragma unroll
  for (int j = 0; j < 8; ++j) oacc[j] = zero;
  float mrun[8], lrun[8];
#pragma unroll
  for (int r = 0; r < 8; ++r) { mrun[r] = -3.0e38f; lrun[r] = 0.0f; }

  const __bf16* Kbase = Kb + ((size_t)b * SP) * DIM_C + h * D_H;
  const __bf16* Vbase = Vt + ((size_t)((b * H_N + h) * D_H)) * SP;

  // issue K/V tile copies for chunk starting at c0 into buffer bufi
  auto issue_tiles = [&](int c0, int bufi) {
#pragma unroll
    for (int it = 0; it < 4; ++it) {
      int c = tid + it * 128;  // 512 x 16B chunks per tile
      CP16(Kbase + (size_t)(c0 + (c >> 4)) * DIM_C + ((c & 15) * 8),
           tk[bufi] + c * 8);
      CP16(Vbase + (size_t)(c >> 2) * SP + c0 + ((c & 3) * 8),
           tv[bufi] + c * 8);
    }
  };

  issue_tiles(0, 0);
  int buf = 0;
  for (int c0 = 0; c0 < SP; c0 += 32) {
    CP_WAIT();
    __syncthreads();
    if (c0 + 32 < SP) issue_tiles(c0 + 32, buf ^ 1);

    // S = Q K^T for 32 keys (2 fragments of 16)
    v8f s0 = zero, s1 = zero;
#pragma unroll
    for (int kk = 0; kk < 4; ++kk) {
      s0 = wmma_bf16(qf[kk], frag_b(tk[buf] + kk * 32, lane, 128), s0);
      s1 = wmma_bf16(qf[kk], frag_b(tk[buf] + 16 * 128 + kk * 32, lane, 128), s1);
    }

    float f0 = (c0 >= 2112) ? ratio * scl : scl;
    float f1 = (c0 + 16 >= 2112) ? ratio * scl : scl;
    bool ok0 = (c0 < SKV), ok1 = (c0 + 16 < SKV);

#pragma unroll
    for (int r = 0; r < 8; ++r) {
      float a0 = ok0 ? s0[r] * f0 : -3.0e38f;
      float a1 = ok1 ? s1[r] * f1 : -3.0e38f;
      float mx = redmax16(fmaxf(a0, a1));
      float mnew = fmaxf(mrun[r], mx);
      float corr = __expf(mrun[r] - mnew);
      float p0 = __expf(a0 - mnew);
      float p1 = __expf(a1 - mnew);
      float rs = redsum16(p0 + p1);
      lrun[r] = lrun[r] * corr + rs;
      mrun[r] = mnew;
#pragma unroll
      for (int j = 0; j < 8; ++j) oacc[j][r] *= corr;
      int pm = r + 8 * (lane >> 4);
      pl[wv][pm * 32 + (lane & 15)] = (__bf16)p0;
      pl[wv][pm * 32 + 16 + (lane & 15)] = (__bf16)p1;
    }
    // O += P @ V   (same-wave LDS RAW: DS ops are in-order per wave)
    v16bf pf = frag_a(pl[wv], lane, 32);
#pragma unroll
    for (int j = 0; j < 8; ++j)
      oacc[j] = wmma_bf16(pf, frag_b(tv[buf] + j * 16 * 32, lane, 32), oacc[j]);
    buf ^= 1;
  }

#pragma unroll
  for (int r = 0; r < 8; ++r) {
    float inv = 1.0f / lrun[r];
    int rowg = q0 + wv * 16 + r + 8 * (lane >> 4);
    __bf16* orow = Ob + ((size_t)(b * SP + rowg)) * DIM_C + h * D_H;
#pragma unroll
    for (int j = 0; j < 8; ++j)
      orow[j * 16 + (lane & 15)] = (__bf16)(oacc[j][r] * inv);
  }
}

// ---------------------------------------------------------------------------
// LayerNorm: y (f32, in d_out) -> yn bf16 in ws (padded layout)
// ---------------------------------------------------------------------------
__global__ __launch_bounds__(256) void ln_kernel(
    const float* __restrict__ y, const float* __restrict__ g,
    const float* __restrict__ be, __bf16* __restrict__ ynb) {
  int row = blockIdx.x;  // 0..8191
  int lane = threadIdx.x & 31, wv = threadIdx.x >> 5;
  const float* yr = y + (size_t)row * DIM_C;
  float s = 0.f, s2 = 0.f;
  for (int i = threadIdx.x; i < DIM_C; i += 256) {
    float v = yr[i];
    s += v; s2 += v * v;
  }
#pragma unroll
  for (int off = 16; off >= 1; off >>= 1) {
    s += __shfl_xor(s, off);
    s2 += __shfl_xor(s2, off);
  }
  __shared__ float rs[8], rs2[8];
  if (lane == 0) { rs[wv] = s; rs2[wv] = s2; }
  __syncthreads();
  float ts = 0.f, ts2 = 0.f;
#pragma unroll
  for (int w = 0; w < 8; ++w) { ts += rs[w]; ts2 += rs2[w]; }
  float mu = ts * (1.0f / DIM_C);
  float var = ts2 * (1.0f / DIM_C) - mu * mu;
  float rstd = rsqrtf(var + 1e-5f);
  int bb = row >> 11, tt = row & 2047;
  __bf16* orow = ynb + ((size_t)(bb * SP + tt)) * DIM_C;
  for (int i = threadIdx.x; i < DIM_C; i += 256)
    orow[i] = (__bf16)((yr[i] - mu) * rstd * g[i] + be[i]);
}

// ---------------------------------------------------------------------------
extern "C" void kernel_launch(void* const* d_in, const int* in_sizes, int n_in,
                              void* d_out, int out_size, void* d_ws, size_t ws_size,
                              hipStream_t stream) {
  const float* x   = (const float*)d_in[0];
  const float* ht  = (const float*)d_in[1];
  const float* ha  = (const float*)d_in[2];
  const float* pp  = (const float*)d_in[3];
  const float* Wq  = (const float*)d_in[4];
  const float* bq  = (const float*)d_in[5];
  const float* Wk  = (const float*)d_in[6];
  const float* bk  = (const float*)d_in[7];
  const float* Wv  = (const float*)d_in[8];
  const float* bv  = (const float*)d_in[9];
  const float* Wo  = (const float*)d_in[10];
  const float* bo  = (const float*)d_in[11];
  const float* lg  = (const float*)d_in[12];
  const float* lb  = (const float*)d_in[13];
  const float* Wf  = (const float*)d_in[14];
  const float* bfb = (const float*)d_in[15];
  const float* gat = (const float*)d_in[16];
  float* out = (float*)d_out;

  char* w = (char*)d_ws;
  const size_t matb = (size_t)MTOT * DIM_C * sizeof(__bf16);   // 17.8 MB
  const size_t wtb  = (size_t)DIM_C * DIM_C * sizeof(__bf16);  // 2 MB
  __bf16* XC  = (__bf16*)w; w += matb;
  __bf16* Qb  = (__bf16*)w; w += matb;
  __bf16* Kb  = (__bf16*)w; w += matb;
  __bf16* Vt  = (__bf16*)w; w += matb;
  __bf16* Ob  = (__bf16*)w; w += matb;
  __bf16* ynb = (__bf16*)w; w += matb;
  __bf16* WqT = (__bf16*)w; w += wtb;
  __bf16* WkT = (__bf16*)w; w += wtb;
  __bf16* WvT = (__bf16*)w; w += wtb;
  __bf16* WoT = (__bf16*)w; w += wtb;
  __bf16* WfT = (__bf16*)w; w += wtb;

  dim3 blk256(256), blk128(128);
  dim3 gwt(32, 32), ggemm(8, MTOT / 128), gattn(T_SZ / 64, H_N, B_SZ);

  pack_kernel<<<MTOT, blk256, 0, stream>>>(x, ht, ha, pp, XC);
  wt_kernel<<<gwt, blk256, 0, stream>>>(Wq, WqT);
  wt_kernel<<<gwt, blk256, 0, stream>>>(Wk, WkT);
  wt_kernel<<<gwt, blk256, 0, stream>>>(Wv, WvT);
  wt_kernel<<<gwt, blk256, 0, stream>>>(Wo, WoT);
  wt_kernel<<<gwt, blk256, 0, stream>>>(Wf, WfT);

  gemm_kernel<<<ggemm, blk256, 0, stream>>>(XC, WqT, bq, 0, Qb, nullptr, nullptr);
  gemm_kernel<<<ggemm, blk256, 0, stream>>>(XC, WkT, bk, 0, Kb, nullptr, nullptr);
  gemm_kernel<<<ggemm, blk256, 0, stream>>>(XC, WvT, bv, 1, Vt, nullptr, nullptr);

  attn_kernel<<<gattn, blk128, 0, stream>>>(Qb, Kb, Vt, Ob, gat);

  // y = Ob @ Wo + bo + x  -> d_out (temporary f32 buffer)
  gemm_kernel<<<ggemm, blk256, 0, stream>>>(Ob, WoT, bo, 2, nullptr, out, x);
  ln_kernel<<<B_SZ * T_SZ, blk256, 0, stream>>>(out, lg, lb, ynb);
  // out = relu(yn @ Wf + bf) -> overwrites d_out
  gemm_kernel<<<ggemm, blk256, 0, stream>>>(ynb, WfT, bfb, 3, nullptr, out, nullptr);
}
